// Linear4Bit_16965120819567
// MI455X (gfx1250) — matrix operand order repaired and checked by reading, not compiled, not run on hardware
//
#include <hip/hip_runtime.h>

// ---------------------------------------------------------------------------
// 4-bit-quantized linear layer for MI455X (gfx1250, wave32, WMMA).
// Y[16384,4096] = X[16384,4096] * dequant(W4)[4096,4096]^T + bias
// Compute-bound (AI ~965 FLOP/B) -> bf16 WMMA 16x16x32, f32 accumulate.
// v4: TILE_M=256 (halves dequant redundancy, 16 WMMA/wave/K-step),
//     v_perm-based f32->bf16 truncation for X (1 VALU per 2 elements),
//     v_cvt_pk_bf16_f32 packed converts for W when available,
//     double-buffered LDS (48 KB), one barrier per 32-deep K step.
// ---------------------------------------------------------------------------

typedef __attribute__((ext_vector_type(16))) __bf16 v16bf;
typedef __attribute__((ext_vector_type(2)))  __bf16 v2bf;
typedef __attribute__((ext_vector_type(8)))  float  v8f;

#define TOKENS  16384
#define IN_F    4096
#define OUT_F   4096
#define NGROUPS_PER_ROW (IN_F / 16)   // 256
#define KSTEP   32
#define TILE_M  256
#define TILE_N  128
#define NKT     (IN_F / KSTEP)        // 128

union FragU {
    v16bf bf;
    uint4 q[2];
};

__device__ __forceinline__ unsigned short f32_to_bf16(float f) {
    return __builtin_bit_cast(unsigned short, (__bf16)f);
}

// pack two f32 -> packed bf16 pair (lo in [15:0], hi in [31:16])
__device__ __forceinline__ unsigned int pack_bf16_pair(float lo, float hi) {
#if __has_builtin(__builtin_amdgcn_cvt_pk_bf16_f32)
    return __builtin_bit_cast(unsigned int, __builtin_amdgcn_cvt_pk_bf16_f32(lo, hi));
#else
    return (unsigned int)f32_to_bf16(lo) | ((unsigned int)f32_to_bf16(hi) << 16);
#endif
}

// truncate two f32 -> packed bf16 pair with a single v_perm_b32
__device__ __forceinline__ unsigned int trunc_bf16_pair(float lo, float hi) {
    return __builtin_amdgcn_perm(__builtin_bit_cast(unsigned int, hi),
                                 __builtin_bit_cast(unsigned int, lo),
                                 0x07060302u);
}

__global__ __launch_bounds__(256)
void linear4bit_wmma_kernel(const float* __restrict__ x,
                            const int* __restrict__ wq4,        // (NGROUPS, 8) byte-per-int
                            const _Float16* __restrict__ wnorm, // (NGROUPS)
                            const float* __restrict__ bias,     // (OUT_F)
                            float* __restrict__ out)
{
    // double-buffered bf16 staging tiles: 2*(16KB + 8KB) = 48 KB LDS
    __shared__ unsigned short Xs[2][TILE_M][KSTEP];   // [m][k]  A operand source
    __shared__ unsigned short Ws[2][KSTEP][TILE_N];   // [k][n]  B operand (W^T) source

    const int t    = threadIdx.x;
    const int lane = t & 31;
    const int wave = t >> 5;

    const int m0 = blockIdx.y * TILE_M;
    const int n0 = blockIdx.x * TILE_N;

    // wave sub-tile: wm (0..3) -> four 16-row blocks, wn (0..1) -> four 16-col blocks
    const int wm = wave >> 1;
    const int wn = wave & 1;

    v8f acc[4][4];
#pragma unroll
    for (int i = 0; i < 4; ++i)
#pragma unroll
        for (int j = 0; j < 4; ++j)
            acc[i][j] = (v8f){0.f, 0.f, 0.f, 0.f, 0.f, 0.f, 0.f, 0.f};

    // per-lane bias for each of the wave's four n16 blocks (N = lane&15 within block)
    float bvals[4];
#pragma unroll
    for (int j = 0; j < 4; ++j)
        bvals[j] = bias[n0 + wn * 64 + j * 16 + (lane & 15)];

    // A-fragment addressing (wave32 16-bit A layout):
    //  lanes 0-15 : row = lane,      K chunks {0..7, 16..23}
    //  lanes 16-31: row = lane - 16, K chunks {8..15, 24..31}
    const int arow  = lane & 15;
    const int akoff = (lane >= 16) ? 8 : 0;
    // B-fragment addressing: N = lane&15, K chunks same as A ({0..7,16..23} / {8..15,24..31})

    // W loader (threads 0..127 only): 2 groups (same k-range, n and n+1) per thread
    const int wqtr = (t >> 6) & 1;   // 0..1 : which 16-k half of the 32-k step
    const int wnp  = (t & 63) * 2;   // even n within tile

    // ---- stage one 32-deep K-step of X and dequantized W^T into buffer `buf` ----
    auto stage = [&](int buf, int k0) {
        // X tile: 256 x 32, fp32 -> bf16 (one full row per thread, v_perm truncation)
        {
            const float* src = x + (size_t)(m0 + t) * IN_F + k0;
#pragma unroll
            for (int c = 0; c < 2; ++c) {
                const float4* s4 = (const float4*)(src + c * 16);
                float fv[16];
#pragma unroll
                for (int i = 0; i < 4; ++i)
                    *(float4*)&fv[4 * i] = s4[i];
                unsigned int p[8];
#pragma unroll
                for (int i = 0; i < 8; ++i)
                    p[i] = trunc_bf16_pair(fv[2 * i], fv[2 * i + 1]);
                uint4* dst = (uint4*)&Xs[buf][t][c * 16];
                dst[0] = make_uint4(p[0], p[1], p[2], p[3]);
                dst[1] = make_uint4(p[4], p[5], p[6], p[7]);
            }
        }
        // W^T tile: 256 groups, threads 0..127 dequantize 2 groups each
        // (adjacent n pair), packed b32 stores. Wave-uniform branch (waves 0-3).
        if (t < 128) {
            const size_t grp0 = (size_t)(n0 + wnp) * NGROUPS_PER_ROW + (k0 >> 4) + wqtr;
            const size_t grp1 = grp0 + NGROUPS_PER_ROW;   // row n+1, same k group
            const int4* qp0 = (const int4*)(wq4 + grp0 * 8);
            const int4* qp1 = (const int4*)(wq4 + grp1 * 8);
            const int4 a0 = qp0[0], b0 = qp0[1];
            const int4 a1 = qp1[0], b1 = qp1[1];
            const float nrm0 = (float)wnorm[grp0];
            const float nrm1 = (float)wnorm[grp1];
            const float sc0  = nrm0 * (2.0f / 15.0f);
            const float sc1  = nrm1 * (2.0f / 15.0f);
            const int qv0[8] = {a0.x, a0.y, a0.z, a0.w, b0.x, b0.y, b0.z, b0.w};
            const int qv1[8] = {a1.x, a1.y, a1.z, a1.w, b1.x, b1.y, b1.z, b1.w};
            const int kbase = wqtr * 16;
#pragma unroll
            for (int j = 0; j < 8; ++j) {
                const int c0 = qv0[j];
                const int c1 = qv1[j];
                const float w0l = fmaf((float)(c0 & 15),  sc0, -nrm0);
                const float w0h = fmaf((float)(c0 >> 4),  sc0, -nrm0);
                const float w1l = fmaf((float)(c1 & 15),  sc1, -nrm1);
                const float w1h = fmaf((float)(c1 >> 4),  sc1, -nrm1);
                *(unsigned int*)&Ws[buf][kbase + 2 * j][wnp]     = pack_bf16_pair(w0l, w1l);
                *(unsigned int*)&Ws[buf][kbase + 2 * j + 1][wnp] = pack_bf16_pair(w0h, w1h);
            }
        }
    };

    // ---- consume buffer `buf`: 8 ds_load_b128 A loads + 64 16-bit B loads + 16 WMMA ----
    auto compute = [&](int buf) {
        FragU afr[4], bfr[4];
#pragma unroll
        for (int i = 0; i < 4; ++i) {
            const int m = (wm * 4 + i) * 16 + arow;
            afr[i].q[0] = *(const uint4*)&Xs[buf][m][akoff];
            afr[i].q[1] = *(const uint4*)&Xs[buf][m][akoff + 16];
        }
#pragma unroll
        for (int j = 0; j < 4; ++j) {
            const int n = wn * 64 + j * 16;
#pragma unroll
            for (int s = 0; s < 16; ++s)
                bfr[j].bf[s] = __builtin_bit_cast(__bf16, Ws[buf][(s < 8) ? (akoff + s) : (akoff + 8 + s)][n + arow]);
        }
#pragma unroll
        for (int i = 0; i < 4; ++i)
#pragma unroll
            for (int j = 0; j < 4; ++j)
                acc[i][j] = __builtin_amdgcn_wmma_f32_16x16x32_bf16(
                    /*neg_a=*/false, afr[i].bf,
                    /*neg_b=*/false, bfr[j].bf,
                    /*c_mod=*/(short)0, acc[i][j],
                    /*reuse_a=*/false, /*reuse_b=*/false);
    };

    // ---- software-pipelined K loop: one barrier per 32-deep K-step ----
    int p = 0;
    stage(0, 0);
    for (int kt = 1; kt < NKT; ++kt) {
        __syncthreads();              // buffer p fully staged / p^1 fully consumed
        stage(p ^ 1, kt * KSTEP);     // overlaps with WMMA on buffer p
        compute(p);
        p ^= 1;
    }
    __syncthreads();
    compute(p);

    // --------------------------- bias add + store -----------------------------
    // D layout: VGPR r -> M = r (lanes 0-15) / r+8 (lanes 16-31), N = lane&15
    const int mlane = (lane >= 16) ? 8 : 0;
#pragma unroll
    for (int i = 0; i < 4; ++i) {
#pragma unroll
        for (int j = 0; j < 4; ++j) {
            const int mbase = m0 + (wm * 4 + i) * 16 + mlane;
            const int n     = n0 + wn * 64 + j * 16 + (lane & 15);
#pragma unroll
            for (int r = 0; r < 8; ++r)
                out[(size_t)(mbase + r) * OUT_F + n] = acc[i][j][r] + bvals[j];
        }
    }
}

extern "C" void kernel_launch(void* const* d_in, const int* in_sizes, int n_in,
                              void* d_out, int out_size, void* d_ws, size_t ws_size,
                              hipStream_t stream) {
    const float*    x     = (const float*)d_in[0];
    const int*      wq4   = (const int*)d_in[1];
    const _Float16* wnorm = (const _Float16*)d_in[2];
    const float*    bias  = (const float*)d_in[3];
    float*          out   = (float*)d_out;

    dim3 grid(OUT_F / TILE_N, TOKENS / TILE_M);  // (32, 64)
    dim3 block(256);
    linear4bit_wmma_kernel<<<grid, block, 0, stream>>>(x, wq4, wnorm, bias, out);
}
